// MyTransformer_87720412054027
// MI455X (gfx1250) — compile-verified, matrix operand order checked by reading, not run on hardware
//
#include <hip/hip_runtime.h>
#include <math.h>
#include <stdint.h>

// ---------------- model constants (match reference) ----------------
#define VOCAB 10000
#define NLAYER 12
#define NHEAD 12
#define CEMB 768
#define TEMB 64
#define WTEDIM (CEMB - TEMB)   // 704
#define BSZ 4
#define TSEQ 1024
#define DHEAD (CEMB / NHEAD)   // 64
#define MROWS (BSZ * TSEQ)     // 4096

typedef __attribute__((ext_vector_type(16))) _Float16 v16h;
typedef __attribute__((ext_vector_type(8)))  _Float16 v8h;
typedef __attribute__((ext_vector_type(8)))  float    v8f;

// ---------------- WMMA helpers ----------------
__device__ __forceinline__ v8f wmma_f16(v16h a, v16h b, v8f c) {
  // D = A(16x32 f16) * B(32x16 f16) + C(16x16 f32)
  return __builtin_amdgcn_wmma_f32_16x16x32_f16(
      /*neg_a=*/false, a, /*neg_b=*/false, b,
      /*c_mod=*/(short)0, c, /*reuse_a=*/false, /*reuse_b=*/false);
}

// A-fragment (and B-fragment via W rows == B columns), row-major source.
// ISA 16-bit A 16x32 layout: lanes 0-15 row M=lane, K 0-7/16-23;
// lanes 16-31 row M=lane-16, K 8-15/24-31.  Two 16B loads per lane.
__device__ __forceinline__ v16h load_frag_rm(const _Float16* __restrict__ base,
                                             int ld, int row, int kbase) {
  int lane = threadIdx.x & 31;
  const _Float16* p = base + (size_t)(row + (lane & 15)) * (size_t)ld
                           + kbase + ((lane >> 4) << 3);
  v8h lo = *(const v8h*)(p);        // K = kbase + 8*half + 0..7
  v8h hi = *(const v8h*)(p + 16);   // K = kbase + 8*half + 16..23
  return __builtin_shufflevector(lo, hi, 0,1,2,3,4,5,6,7,8,9,10,11,12,13,14,15);
}

// B-fragment from a K-major (strided) source: element [k][n] = base[k*ld + n].
__device__ __forceinline__ v16h load_frag_b_strided(const _Float16* __restrict__ base,
                                                    int ld, int k0, int n0) {
  int lane = threadIdx.x & 31;
  int n = n0 + (lane & 15);
  int khi = (lane >> 4) << 3;
  v16h r;
#pragma unroll
  for (int e = 0; e < 16; ++e) {
    int k = k0 + (e & 7) + ((e >> 3) << 4) + khi;
    r[e] = base[(size_t)k * (size_t)ld + n];
  }
  return r;
}

// ---------------- CDNA5 async Global->LDS (ASYNCcnt-tracked) ----------------
__device__ __forceinline__ void async_copy16(uint32_t lds_off, const _Float16* gaddr) {
  // GLOBAL_LOAD_ASYNC_TO_LDS_B128 (GV mode): vdst = LDS byte addr, vaddr = 64b global
  asm volatile("global_load_async_to_lds_b128 %0, %1, off"
               :: "v"(lds_off), "v"(gaddr) : "memory");
}
#define WAIT_ASYNCCNT_0() asm volatile("s_wait_asynccnt 0x0" ::: "memory")
#define WAIT_ASYNCCNT_3() asm volatile("s_wait_asynccnt 0x3" ::: "memory")

// ---------------- elementwise / setup kernels ----------------
__global__ void f32_to_f16_kernel(const float* __restrict__ in,
                                  _Float16* __restrict__ out, size_t n) {
  size_t i = (size_t)blockIdx.x * blockDim.x + threadIdx.x;
  if (i < n) out[i] = (_Float16)in[i];
}

__global__ void embed_kernel(const int* __restrict__ idx, const float* __restrict__ ts,
                             const float* __restrict__ wte, const float* __restrict__ freq,
                             const float* __restrict__ phase, const float* __restrict__ wpe,
                             float* __restrict__ x) {
  int bt = blockIdx.x;
  int t = bt % TSEQ;
  int id = idx[bt];
  float tv = ts[bt];
  for (int c = threadIdx.x; c < CEMB; c += blockDim.x) {
    float v;
    if (c < WTEDIM) v = wte[(size_t)id * WTEDIM + c];
    else { int e = c - WTEDIM; v = cosf(tv * freq[e] + phase[e]); }
    x[(size_t)bt * CEMB + c] = v + wpe[(size_t)t * CEMB + c];
  }
}

__global__ void ln_f16_kernel(const float* __restrict__ x, const float* __restrict__ g,
                              const float* __restrict__ b, _Float16* __restrict__ out, int C) {
  __shared__ float s1[256], s2[256];
  int row = blockIdx.x;
  const float* xr = x + (size_t)row * C;
  float ls = 0.f, lq = 0.f;
  for (int c = threadIdx.x; c < C; c += blockDim.x) { float v = xr[c]; ls += v; lq += v * v; }
  s1[threadIdx.x] = ls; s2[threadIdx.x] = lq;
  __syncthreads();
  for (int st = 128; st > 0; st >>= 1) {
    if ((int)threadIdx.x < st) { s1[threadIdx.x] += s1[threadIdx.x + st];
                                 s2[threadIdx.x] += s2[threadIdx.x + st]; }
    __syncthreads();
  }
  float mu = s1[0] / (float)C;
  float var = s2[0] / (float)C - mu * mu;
  float rs = rsqrtf(var + 1e-5f);
  for (int c = threadIdx.x; c < C; c += blockDim.x)
    out[(size_t)row * C + c] = (_Float16)((xr[c] - mu) * rs * g[c] + b[c]);
}

__global__ void tpred_kernel(const _Float16* __restrict__ h, const float* __restrict__ w,
                             float* __restrict__ out) {
  __shared__ float s[256];
  int row = blockIdx.x;
  float acc = 0.f;
  for (int c = threadIdx.x; c < CEMB; c += blockDim.x)
    acc += (float)h[(size_t)row * CEMB + c] * w[c];
  s[threadIdx.x] = acc;
  __syncthreads();
  for (int st = 128; st > 0; st >>= 1) {
    if ((int)threadIdx.x < st) s[threadIdx.x] += s[threadIdx.x + st];
    __syncthreads();
  }
  if (threadIdx.x == 0) out[row] = s[0];
}

// ---------------- WMMA GEMM with async Global->LDS double-buffered staging ----
// out[M,N] = A[M,K](f16) * W[N,K]^T(f16) + bias
// mode 0: f16 out; 1: f32 out + residual; 2: f16 out + exact GELU; 3: f32 out plain
// Block tile: 128(M) x 64(N), K-slab 32.  8 waves: 4 along M x 2 along N, 32x32 each.
__global__ __launch_bounds__(256)
void wmma_gemm_kernel(const _Float16* __restrict__ A, const _Float16* __restrict__ W,
                      const float* __restrict__ bias, const float* __restrict__ resid,
                      float* __restrict__ outF32, _Float16* __restrict__ outF16,
                      int M, int N, int K, int mode) {
  __shared__ _Float16 As[2][128 * 32];   // 2 x 8KB
  __shared__ _Float16 Bs[2][64 * 32];    // 2 x 4KB
  int tid  = threadIdx.x;
  int wave = tid >> 5;
  int lane = tid & 31;
  int rowB = blockIdx.y * 128;                 // block M origin
  int colB = blockIdx.x * 64;                  // block N origin
  int rowW = (wave & 3) * 32;                  // wave M offset inside block
  int colb = (wave >> 2) * 32;                 // wave N offset inside block
  int maxRowW = N - 1;

  // cooperative async stage of one K-slab into LDS buffer `buf`
  auto stage = [&](int kb, int buf) {
    // A tile: 128 rows x 32 halves = 512 x 16B chunks -> 2 per thread
#pragma unroll
    for (int i = 0; i < 2; ++i) {
      int c = tid + i * 256;
      int row = c >> 2, seg = c & 3;
      const _Float16* g = A + (size_t)(rowB + row) * K + kb + seg * 8;
      uint32_t loff = (uint32_t)(size_t)(&As[buf][row * 32 + seg * 8]);
      async_copy16(loff, g);
    }
    // B tile: 64 rows x 32 halves = 256 x 16B chunks -> 1 per thread
    {
      int row = tid >> 2, seg = tid & 3;
      int rw = colB + row; if (rw > maxRowW) rw = maxRowW;  // clamp for ragged N
      const _Float16* g = W + (size_t)rw * K + kb + seg * 8;
      uint32_t loff = (uint32_t)(size_t)(&Bs[buf][row * 32 + seg * 8]);
      async_copy16(loff, g);
    }
  };

  v8f c00 = {}, c01 = {}, c10 = {}, c11 = {};
  int buf = 0;
  stage(0, 0);
  for (int kb = 0; kb < K; kb += 32) {
    int nxt = buf ^ 1;
    bool more = (kb + 32 < K);
    if (more) {
      stage(kb + 32, nxt);                      // overlap next slab's DMA
      if (kb + 64 < K) {                        // warm L2 two slabs ahead
        __builtin_prefetch(A + (size_t)(rowB + (tid >> 2)) * K + kb + 64, 0, 1);
      }
      WAIT_ASYNCCNT_3();                        // current slab landed; next 3 in flight
    } else {
      WAIT_ASYNCCNT_0();
    }
    __syncthreads();                            // all waves' slab data visible

    v16h a0 = load_frag_rm(&As[buf][0], 32, rowW,      0);   // ds_load_b128 path
    v16h a1 = load_frag_rm(&As[buf][0], 32, rowW + 16, 0);
    v16h b0 = load_frag_rm(&Bs[buf][0], 32, colb,      0);
    v16h b1 = load_frag_rm(&Bs[buf][0], 32, colb + 16, 0);
    c00 = wmma_f16(a0, b0, c00);
    c01 = wmma_f16(a0, b1, c01);
    c10 = wmma_f16(a1, b0, c10);
    c11 = wmma_f16(a1, b1, c11);

    __syncthreads();                            // done reading before buf is re-staged
    buf = nxt;
  }

  // epilogue: C layout = N across lanes (lane&15), M = v + 8*(lane>>4)
  int row0 = rowB + rowW;
  int col0 = colB + colb;
  v8f cf[4] = {c00, c01, c10, c11};
  int mof[4] = {0, 0, 16, 16};
  int nof[4] = {0, 16, 0, 16};
#pragma unroll
  for (int f = 0; f < 4; ++f) {
    int n = col0 + nof[f] + (lane & 15);
    if (n >= N) continue;
    int mb = row0 + mof[f] + ((lane >> 4) << 3);
    float bv = bias ? bias[n] : 0.f;
#pragma unroll
    for (int v = 0; v < 8; ++v) {
      float val = cf[f][v] + bv;
      size_t o = (size_t)(mb + v) * (size_t)N + n;
      if (mode == 0) {
        outF16[o] = (_Float16)val;
      } else if (mode == 1) {
        outF32[o] = resid[o] + val;
      } else if (mode == 2) {
        float gl = 0.5f * val * (1.f + erff(val * 0.70710678118654752f)); // exact GELU
        outF16[o] = (_Float16)gl;
      } else {
        outF32[o] = val;
      }
    }
  }
}

// ---------------- WMMA flash attention (one wave / 16-query tile) ----------------
__global__ __launch_bounds__(32)
void attn_kernel(const _Float16* __restrict__ qkv, _Float16* __restrict__ yh) {
  __shared__ _Float16 pstage[16 * 32];  // P tile, A-layout staging (1 KB)
  int qt = blockIdx.x, h = blockIdx.y, b = blockIdx.z;
  int lane = threadIdx.x & 31;
  int half = lane >> 4;
  const int ld3 = 3 * CEMB;
  const _Float16* Qb = qkv + ((size_t)(b * TSEQ) + qt * 16) * ld3 + h * DHEAD;
  const _Float16* Kb = qkv + (size_t)(b * TSEQ) * ld3 + CEMB + h * DHEAD;
  const _Float16* Vb = qkv + (size_t)(b * TSEQ) * ld3 + 2 * CEMB + h * DHEAD;

  v16h qa0 = load_frag_rm(Qb, ld3, 0, 0);    // d 0..31
  v16h qa1 = load_frag_rm(Qb, ld3, 0, 32);   // d 32..63

  v8f o0 = {}, o1 = {}, o2 = {}, o3 = {};
  float mrow[8], lrow[8];
#pragma unroll
  for (int v = 0; v < 8; ++v) { mrow[v] = -1e30f; lrow[v] = 0.f; }
  const float scale = rsqrtf((float)DHEAD);

  int npair = (qt >> 1) + 1;   // 32-key pairs covering tiles 0..qt (tail tile masked)
  for (int p = 0; p < npair; ++p) {
    int kt0 = 2 * p, kt1 = 2 * p + 1;
    // S = Q @ K^T  (reduce over d=64 -> two WMMAs per 16-key tile)
    v8f s0 = {}, s1 = {};
    s0 = wmma_f16(qa0, load_frag_rm(Kb, ld3, kt0 * 16, 0),  s0);
    s0 = wmma_f16(qa1, load_frag_rm(Kb, ld3, kt0 * 16, 32), s0);
    s1 = wmma_f16(qa0, load_frag_rm(Kb, ld3, kt1 * 16, 0),  s1);
    s1 = wmma_f16(qa1, load_frag_rm(Kb, ld3, kt1 * 16, 32), s1);

    int key0 = kt0 * 16 + (lane & 15);
    int key1 = kt1 * 16 + (lane & 15);
    float f0[8], f1[8], p0[8], p1[8];
#pragma unroll
    for (int v = 0; v < 8; ++v) {
      int q = qt * 16 + v + 8 * half;
      f0[v] = (key0 <= q) ? s0[v] * scale : -1e30f;
      f1[v] = (key1 <= q) ? s1[v] * scale : -1e30f;
    }
    // online softmax: row stats live per (lane-half, v); reduce across 16 lanes
#pragma unroll
    for (int v = 0; v < 8; ++v) {
      float mx = fmaxf(f0[v], f1[v]);
      mx = fmaxf(mx, __shfl_xor(mx, 1, 32));
      mx = fmaxf(mx, __shfl_xor(mx, 2, 32));
      mx = fmaxf(mx, __shfl_xor(mx, 4, 32));
      mx = fmaxf(mx, __shfl_xor(mx, 8, 32));
      float mnew = fmaxf(mrow[v], mx);
      float alpha = __expf(mrow[v] - mnew);
      p0[v] = __expf(f0[v] - mnew);
      p1[v] = __expf(f1[v] - mnew);
      float sm = p0[v] + p1[v];
      sm += __shfl_xor(sm, 1, 32);
      sm += __shfl_xor(sm, 2, 32);
      sm += __shfl_xor(sm, 4, 32);
      sm += __shfl_xor(sm, 8, 32);
      lrow[v] = lrow[v] * alpha + sm;
      mrow[v] = mnew;
      o0[v] *= alpha; o1[v] *= alpha; o2[v] *= alpha; o3[v] *= alpha;
    }
    // transpose P from C-layout to A-layout through LDS
#pragma unroll
    for (int v = 0; v < 8; ++v) {
      int r = v + 8 * half;
      pstage[r * 32 + (lane & 15)]      = (_Float16)p0[v];
      pstage[r * 32 + 16 + (lane & 15)] = (_Float16)p1[v];
    }
    __syncthreads();
    v16h pa;
    {
      int r = lane & 15; int kh = half << 3;
      v8h lo = *(const v8h*)&pstage[r * 32 + kh];
      v8h hi = *(const v8h*)&pstage[r * 32 + 16 + kh];
      pa = __builtin_shufflevector(lo, hi, 0,1,2,3,4,5,6,7,8,9,10,11,12,13,14,15);
    }
    __syncthreads();
    // O += P @ V  (32 keys x 64 dims -> 4 WMMAs)
    o0 = wmma_f16(pa, load_frag_b_strided(Vb, ld3, p * 32, 0),  o0);
    o1 = wmma_f16(pa, load_frag_b_strided(Vb, ld3, p * 32, 16), o1);
    o2 = wmma_f16(pa, load_frag_b_strided(Vb, ld3, p * 32, 32), o2);
    o3 = wmma_f16(pa, load_frag_b_strided(Vb, ld3, p * 32, 48), o3);
  }

  // finalize: divide by row sum, write f16 y[b,t, h*D + d]
  v8f oo[4] = {o0, o1, o2, o3};
#pragma unroll
  for (int j = 0; j < 4; ++j) {
    int d = j * 16 + (lane & 15);
#pragma unroll
    for (int v = 0; v < 8; ++v) {
      int m = qt * 16 + v + 8 * half;
      float val = oo[j][v] / lrow[v];
      yh[((size_t)(b * TSEQ) + m) * CEMB + h * DHEAD + d] = (_Float16)val;
    }
  }
}

// ---------------- host orchestration ----------------
static inline void launch_gemm(const _Float16* A, const _Float16* W, const float* bias,
                               const float* resid, float* of32, _Float16* of16,
                               int M, int N, int K, int mode, hipStream_t s) {
  dim3 grid((N + 63) / 64, M / 128);
  wmma_gemm_kernel<<<grid, 256, 0, s>>>(A, W, bias, resid, of32, of16, M, N, K, mode);
}

extern "C" void kernel_launch(void* const* d_in, const int* in_sizes, int n_in,
                              void* d_out, int out_size, void* d_ws, size_t ws_size,
                              hipStream_t stream) {
  (void)in_sizes; (void)n_in; (void)out_size; (void)ws_size;
  const int*   idx     = (const int*)d_in[0];
  const float* ts      = (const float*)d_in[1];
  const float* wte     = (const float*)d_in[2];
  const float* freq    = (const float*)d_in[3];
  const float* phase   = (const float*)d_in[4];
  const float* wpe     = (const float*)d_in[5];
  const float* ln1_w   = (const float*)d_in[6];
  const float* ln1_b   = (const float*)d_in[7];
  const float* qkv_w   = (const float*)d_in[8];
  const float* qkv_b   = (const float*)d_in[9];
  const float* proj_w  = (const float*)d_in[10];
  const float* proj_b  = (const float*)d_in[11];
  const float* ln2_w   = (const float*)d_in[12];
  const float* ln2_b   = (const float*)d_in[13];
  const float* fc_w    = (const float*)d_in[14];
  const float* fc_b    = (const float*)d_in[15];
  const float* fc2_w   = (const float*)d_in[16];
  const float* fc2_b   = (const float*)d_in[17];
  const float* lnf_w   = (const float*)d_in[18];
  const float* lnf_b   = (const float*)d_in[19];
  const float* lm_w    = (const float*)d_in[20];
  const float* tp_w    = (const float*)d_in[21];
  float* out = (float*)d_out;

  // carve workspace
  char* ws = (char*)d_ws;
  auto carve = [&](size_t bytes) -> char* {
    char* p = ws; ws += (bytes + 255) & ~(size_t)255; return p;
  };
  const size_t nWqkv = (size_t)NLAYER * 3 * CEMB * CEMB;
  const size_t nWprj = (size_t)NLAYER * CEMB * CEMB;
  const size_t nWfc  = (size_t)NLAYER * 4 * CEMB * CEMB;
  const size_t nWfc2 = (size_t)NLAYER * CEMB * 4 * CEMB;
  const size_t nWlm  = (size_t)VOCAB * CEMB;
  _Float16* Wqkv = (_Float16*)carve(nWqkv * 2);
  _Float16* Wprj = (_Float16*)carve(nWprj * 2);
  _Float16* Wfc  = (_Float16*)carve(nWfc * 2);
  _Float16* Wfc2 = (_Float16*)carve(nWfc2 * 2);
  _Float16* Wlm  = (_Float16*)carve(nWlm * 2);
  float*    x0   = (float*)   carve((size_t)MROWS * CEMB * 4);
  float*    x1   = (float*)   carve((size_t)MROWS * CEMB * 4);
  _Float16* hln  = (_Float16*)carve((size_t)MROWS * CEMB * 2);
  _Float16* qkvh = (_Float16*)carve((size_t)MROWS * 3 * CEMB * 2);
  _Float16* yh   = (_Float16*)carve((size_t)MROWS * CEMB * 2);
  _Float16* act  = (_Float16*)carve((size_t)MROWS * 4 * CEMB * 2);

  // weight down-conversion (f32 -> f16), deterministic every call
  auto cvt = [&](const float* src, _Float16* dst, size_t n) {
    f32_to_f16_kernel<<<(unsigned)((n + 255) / 256), 256, 0, stream>>>(src, dst, n);
  };
  cvt(qkv_w, Wqkv, nWqkv);
  cvt(proj_w, Wprj, nWprj);
  cvt(fc_w,  Wfc,  nWfc);
  cvt(fc2_w, Wfc2, nWfc2);
  cvt(lm_w,  Wlm,  nWlm);

  // embeddings: tok ++ time-cos + wpe
  embed_kernel<<<MROWS, 256, 0, stream>>>(idx, ts, wte, freq, phase, wpe, x0);

  for (int l = 0; l < NLAYER; ++l) {
    // x -> LN1 -> f16
    ln_f16_kernel<<<MROWS, 256, 0, stream>>>(x0, ln1_w + (size_t)l * CEMB,
                                             ln1_b + (size_t)l * CEMB, hln, CEMB);
    // qkv = LN1(x) @ qkv_w^T + qkv_b   (f16 out)
    launch_gemm(hln, Wqkv + (size_t)l * 3 * CEMB * CEMB,
                qkv_b + (size_t)l * 3 * CEMB, nullptr, nullptr, qkvh,
                MROWS, 3 * CEMB, CEMB, /*mode=*/0, stream);
    // flash attention -> y (f16, heads concatenated)
    attn_kernel<<<dim3(TSEQ / 16, NHEAD, BSZ), 32, 0, stream>>>(qkvh, yh);
    // x = x + y @ proj_w^T + proj_b
    launch_gemm(yh, Wprj + (size_t)l * CEMB * CEMB,
                proj_b + (size_t)l * CEMB, x0, x1, nullptr,
                MROWS, CEMB, CEMB, /*mode=*/1, stream);
    // LN2 -> f16
    ln_f16_kernel<<<MROWS, 256, 0, stream>>>(x1, ln2_w + (size_t)l * CEMB,
                                             ln2_b + (size_t)l * CEMB, hln, CEMB);
    // m = gelu(LN2(x) @ fc_w^T + fc_b)  (f16 out)
    launch_gemm(hln, Wfc + (size_t)l * 4 * CEMB * CEMB,
                fc_b + (size_t)l * 4 * CEMB, nullptr, nullptr, act,
                MROWS, 4 * CEMB, CEMB, /*mode=*/2, stream);
    // x = x + m @ fc2_w^T + fc2_b
    launch_gemm(act, Wfc2 + (size_t)l * CEMB * 4 * CEMB,
                fc2_b + (size_t)l * CEMB, x1, x0, nullptr,
                MROWS, CEMB, 4 * CEMB, /*mode=*/1, stream);
  }

  // final LN, lm_head, time-pred
  ln_f16_kernel<<<MROWS, 256, 0, stream>>>(x0, lnf_w, lnf_b, hln, CEMB);
  launch_gemm(hln, Wlm, nullptr, nullptr, out, nullptr,
              MROWS, VOCAB, CEMB, /*mode=*/3, stream);
  tpred_kernel<<<MROWS, 256, 0, stream>>>(hln, tp_w, out + (size_t)MROWS * VOCAB);
}
